// GCNEncoder_49907519979654
// MI455X (gfx1250) — matrix operand order repaired and checked by reading, not compiled, not run on hardware
//
#include <hip/hip_runtime.h>

// CDNA5 / gfx1250 GCN forward.
// WMMA f16->f32 for dense transforms (B pre-swizzled into fragment layout,
// 4 row-tiles per wave to amortize B); f32 global atomics for the
// L2-resident degree / scatter-add / pool segments.

typedef _Float16 v16h __attribute__((ext_vector_type(16)));
typedef _Float16 v8h  __attribute__((ext_vector_type(8)));
typedef float    v8f  __attribute__((ext_vector_type(8)));

static inline unsigned cdiv_u(long long a, long long b) {
    return (unsigned)((a + b - 1) / b);
}

// ---------------- degree / norm ----------------

__global__ void k_deg_init(float* __restrict__ dinv, int n) {
    int i = blockIdx.x * blockDim.x + threadIdx.x;
    if (i < n) dinv[i] = 1.0f;   // self-loop contributes 1 to every node
}

__global__ void k_deg_edges(const int* __restrict__ dst, float* __restrict__ dinv, int e) {
    int i = blockIdx.x * blockDim.x + threadIdx.x;
    if (i < e) atomicAdd(&dinv[dst[i]], 1.0f);
}

__global__ void k_rsqrt(float* __restrict__ dinv, int n) {
    int i = blockIdx.x * blockDim.x + threadIdx.x;
    if (i < n) dinv[i] = rsqrtf(fmaxf(dinv[i], 1.0f));
}

// norm[e] = dinv[src]*dinv[dst], computed once per edge
__global__ void k_edge_norm(const int* __restrict__ src, const int* __restrict__ dst,
                            const float* __restrict__ dinv, float* __restrict__ nrm, int e) {
    int i = blockIdx.x * blockDim.x + threadIdx.x;
    if (i < e) nrm[i] = dinv[src[i]] * dinv[dst[i]];
}

// ---------------- dense transform prep ----------------

// x [N x 3] f32 -> a16 [N x 32] f16, zero padded
__global__ void k_cast_x(const float* __restrict__ x, _Float16* __restrict__ a16,
                         long long total) {
    long long t = (long long)blockIdx.x * blockDim.x + threadIdx.x;
    if (t >= total) return;
    int i = (int)(t >> 5);
    int k = (int)t & 31;
    a16[t] = (_Float16)((k < 3) ? x[(size_t)i * 3 + k] : 0.0f);
}

// W [K_real x out_real] f32 -> Bf in WMMA B-fragment layout:
//   Bf[((kt*n_tiles + n_tile)*32 + lane)*16 + h]
//   lane&15 = column within tile, lane>>4 selects K range {0..15 | 16..31} of the k-step,
//   h = K offset within that range.
__global__ void k_prep_w_frag(const float* __restrict__ W, _Float16* __restrict__ Bf,
                              int K_real, int out_real, int n_tiles_n, int total) {
    int t = blockIdx.x * blockDim.x + threadIdx.x;
    if (t >= total) return;
    int h       = t & 15;
    int lane    = (t >> 4) & 31;
    int tileidx = t >> 9;
    int n_tile  = tileidx % n_tiles_n;
    int kt      = tileidx / n_tiles_n;
    int K   = (kt << 5) + ((lane >> 4) << 4) + h;
    int col = (n_tile << 4) + (lane & 15);
    Bf[t] = (_Float16)((K < K_real && col < out_real) ? W[(size_t)K * out_real + col] : 0.0f);
}

// ---------------- WMMA matmul: C[N x out] = A[N x K_pad] @ W ----------------
// One wave handles 4 consecutive 16-row tiles for one 16-col tile; B fragments
// (KSTEPS of them) are loaded once into registers and reused 4x.

template <int KSTEPS>
__global__ void k_matmul_wmma(const _Float16* __restrict__ A,
                              const _Float16* __restrict__ Bf,
                              float* __restrict__ C,
                              int n_tiles_n, int out_stride, int out_cols,
                              int n_rows, int row_groups) {
    int wave = (int)((blockIdx.x * (long long)blockDim.x + threadIdx.x) >> 5);
    if (wave >= row_groups * n_tiles_n) return;  // wave-uniform
    const int K_pad = KSTEPS * 32;
    int lane   = threadIdx.x & 31;
    int rg     = wave / n_tiles_n;
    int n_tile = wave % n_tiles_n;
    int m      = lane & 15;
    int hi8    = (lane >> 4) * 8;
    int col    = (n_tile << 4) + m;

    // B fragments: two b128 loads each, contiguous per lane
    v16h b[KSTEPS];
#pragma unroll
    for (int kt = 0; kt < KSTEPS; ++kt) {
        const _Float16* bp = Bf + ((size_t)((kt * n_tiles_n + n_tile) * 32 + lane) << 4);
        v8h blo = *(const v8h*)(bp);
        v8h bhi = *(const v8h*)(bp + 8);
#pragma unroll
        for (int i = 0; i < 8; ++i) { b[kt][i] = blo[i]; b[kt][8 + i] = bhi[i]; }
    }

#pragma unroll
    for (int rt = 0; rt < 4; ++rt) {
        int row_base = ((rg << 2) + rt) << 4;
        if (row_base >= n_rows) break;           // wave-uniform
        const _Float16* rowp = A + (size_t)(row_base + m) * K_pad;
        v8f acc = {};
#pragma unroll
        for (int kt = 0; kt < KSTEPS; ++kt) {
            // A k order used here (contiguous): lanes 0-15 = K k0+0..15; lanes 16-31 = K k0+16..31 (matches Bf)
            v8h alo = *(const v8h*)(rowp + kt * 32 + 2 * hi8);
            v8h ahi = *(const v8h*)(rowp + kt * 32 + 2 * hi8 + 8);
            v16h a;
#pragma unroll
            for (int i = 0; i < 8; ++i) { a[i] = alo[i]; a[8 + i] = ahi[i]; }
            acc = __builtin_amdgcn_wmma_f32_16x16x32_f16(
                false, a, false, b[kt], (short)0, acc, false, false);
        }
        if (col < out_cols) {
#pragma unroll
            for (int r = 0; r < 8; ++r) {
                int mm = r + hi8;                // C layout: VGPR r -> M=r (+8 for hi half)
                C[(size_t)(row_base + mm) * out_stride + col] = acc[r];
            }
        }
    }
}

// ---------------- aggregation ----------------

// agg = dinv[i]^2 * hw  (self-loop term, also initializes agg)
__global__ void k_self_init(const float* __restrict__ dinv, const float* __restrict__ hw,
                            float* __restrict__ agg, long long total, int out_shift) {
    long long t = (long long)blockIdx.x * blockDim.x + threadIdx.x;
    if (t >= total) return;
    int i = (int)(t >> out_shift);
    float di = dinv[i];
    agg[t] = di * di * hw[t];
}

// scatter-add: agg[dst] += norm[e] * hw[src]; one thread = VEC channels of one edge
template <int VEC>
__global__ void k_edge_scatter(const int* __restrict__ src, const int* __restrict__ dst,
                               const float* __restrict__ nrm,
                               const float* __restrict__ hw, float* __restrict__ agg,
                               long long total, int pe_shift, int out) {
    long long t = (long long)blockIdx.x * blockDim.x + threadIdx.x;
    if (t >= total) return;
    int e = (int)(t >> pe_shift);
    int c = ((int)t & ((1 << pe_shift) - 1)) * VEC;
    int s = src[e], d = dst[e];
    float w = nrm[e];
    const float* hp = hw + (size_t)s * out + c;
    float* ap = agg + (size_t)d * out + c;
    if (VEC == 4) {
        float4 hv = *(const float4*)hp;          // global_load_b128
        atomicAdd(ap + 0, w * hv.x);
        atomicAdd(ap + 1, w * hv.y);
        atomicAdd(ap + 2, w * hv.z);
        atomicAdd(ap + 3, w * hv.w);
    } else {
        float2 hv = *(const float2*)hp;          // global_load_b64
        atomicAdd(ap + 0, w * hv.x);
        atomicAdd(ap + 1, w * hv.y);
    }
}

// h_next(f16) = relu(agg + b), written in padded f16 layout for next WMMA
__global__ void k_bias_relu_cast(const float* __restrict__ agg, const float* __restrict__ b,
                                 _Float16* __restrict__ a16, long long total, int out_shift) {
    long long t = (long long)blockIdx.x * blockDim.x + threadIdx.x;
    if (t >= total) return;
    int c = (int)t & ((1 << out_shift) - 1);
    float v = agg[t] + b[c];
    a16[t] = (_Float16)fmaxf(v, 0.0f);
}

// ---------------- global mean pool ----------------

__global__ void k_pool_zero(float* __restrict__ pool, float* __restrict__ cnt) {
    int t = threadIdx.x;
    if (t < 256) pool[t] = 0.0f;
    if (t < 128) cnt[t] = 0.0f;
}

__global__ void k_pool_acc(const float* __restrict__ h3, const int* __restrict__ batch,
                           float* __restrict__ pool, float* __restrict__ cnt, int n) {
    int i = blockIdx.x * blockDim.x + threadIdx.x;
    if (i >= n) return;
    int g = batch[i];
    float2 hv = *(const float2*)(h3 + (size_t)i * 2);
    atomicAdd(&pool[g * 2 + 0], hv.x);
    atomicAdd(&pool[g * 2 + 1], hv.y);
    atomicAdd(&cnt[g], 1.0f);
}

// out = (sum + cnt*b3) / max(cnt,1)  == mean(agg)+b3 when cnt>0, 0 when cnt==0
__global__ void k_pool_final(const float* __restrict__ pool, const float* __restrict__ cnt,
                             const float* __restrict__ b3, float* __restrict__ out) {
    int t = threadIdx.x;
    if (t >= 256) return;
    int g = t >> 1, c = t & 1;
    float cn = cnt[g];
    out[t] = (pool[t] + cn * b3[c]) / fmaxf(cn, 1.0f);
}

// ---------------- host launch ----------------

extern "C" void kernel_launch(void* const* d_in, const int* in_sizes, int n_in,
                              void* d_out, int out_size, void* d_ws, size_t ws_size,
                              hipStream_t stream) {
    const float* x     = (const float*)d_in[0];
    const int*   ei    = (const int*)  d_in[1];
    const int*   batch = (const int*)  d_in[2];
    const float* W1    = (const float*)d_in[3];
    const float* b1    = (const float*)d_in[4];
    const float* W2    = (const float*)d_in[5];
    const float* b2    = (const float*)d_in[6];
    const float* W3    = (const float*)d_in[7];
    const float* b3    = (const float*)d_in[8];
    float* out = (float*)d_out;

    const int N = in_sizes[0] / 3;
    const int E = in_sizes[1] / 2;
    const int* src = ei;
    const int* dst = ei + E;

    // workspace carve-up (~77 MB)
    char* ws = (char*)d_ws;
    size_t off = 0;
    auto carve = [&](size_t bytes) -> void* {
        void* p = ws + off;
        off = (off + bytes + 255) & ~(size_t)255;
        return p;
    };
    float*    dinv = (float*)   carve((size_t)N * 4);
    float*    nrm  = (float*)   carve((size_t)E * 4);
    float*    hw   = (float*)   carve((size_t)N * 64 * 4);
    float*    agg  = (float*)   carve((size_t)N * 64 * 4);
    _Float16* a16  = (_Float16*)carve((size_t)N * 64 * 2);
    _Float16* Bf   = (_Float16*)carve((size_t)2 * 4 * 512 * 2);  // <= 2 ksteps * 4 ntiles
    float*    pool = (float*)   carve(256 * 4);
    float*    cnt  = (float*)   carve(128 * 4);
    (void)ws_size; (void)n_in; (void)out_size;

    const int B = 256;

    // degrees -> dinv -> per-edge norm
    k_deg_init <<<cdiv_u(N, B), B, 0, stream>>>(dinv, N);
    k_deg_edges<<<cdiv_u(E, B), B, 0, stream>>>(dst, dinv, E);
    k_rsqrt    <<<cdiv_u(N, B), B, 0, stream>>>(dinv, N);
    k_edge_norm<<<cdiv_u(E, B), B, 0, stream>>>(src, dst, dinv, nrm, E);

    const int row_tiles  = N / 16;                 // 100000 -> 6250 exactly
    const int row_groups = (row_tiles + 3) / 4;    // 1563

    // ---- layer 1: 3 -> 32 (KSTEPS=1, 2 n-tiles) ----
    k_prep_w_frag<<<cdiv_u(1 * 2 * 512, B), B, 0, stream>>>(W1, Bf, 3, 32, 2, 1 * 2 * 512);
    long long totN32 = (long long)N * 32;
    k_cast_x<<<cdiv_u(totN32, B), B, 0, stream>>>(x, a16, totN32);
    k_matmul_wmma<1><<<cdiv_u(row_groups * 2, 4), 128, 0, stream>>>(
        a16, Bf, hw, 2, 32, 32, N, row_groups);
    k_self_init<<<cdiv_u(totN32, B), B, 0, stream>>>(dinv, hw, agg, totN32, 5);
    long long totE8 = (long long)E * 8;
    k_edge_scatter<4><<<cdiv_u(totE8, B), B, 0, stream>>>(src, dst, nrm, hw, agg, totE8, 3, 32);
    k_bias_relu_cast<<<cdiv_u(totN32, B), B, 0, stream>>>(agg, b1, a16, totN32, 5);

    // ---- layer 2: 32 -> 64 (KSTEPS=1, 4 n-tiles) ----
    k_prep_w_frag<<<cdiv_u(1 * 4 * 512, B), B, 0, stream>>>(W2, Bf, 32, 64, 4, 1 * 4 * 512);
    k_matmul_wmma<1><<<cdiv_u(row_groups * 4, 4), 128, 0, stream>>>(
        a16, Bf, hw, 4, 64, 64, N, row_groups);
    long long totN64 = (long long)N * 64;
    k_self_init<<<cdiv_u(totN64, B), B, 0, stream>>>(dinv, hw, agg, totN64, 6);
    long long totE16 = (long long)E * 16;
    k_edge_scatter<4><<<cdiv_u(totE16, B), B, 0, stream>>>(src, dst, nrm, hw, agg, totE16, 4, 64);
    k_bias_relu_cast<<<cdiv_u(totN64, B), B, 0, stream>>>(agg, b2, a16, totN64, 6);

    // ---- layer 3: 64 -> 2 (KSTEPS=2, 1 n-tile, store only 2 cols) ----
    k_prep_w_frag<<<cdiv_u(2 * 1 * 512, B), B, 0, stream>>>(W3, Bf, 64, 2, 1, 2 * 1 * 512);
    k_matmul_wmma<2><<<cdiv_u(row_groups * 1, 4), 128, 0, stream>>>(
        a16, Bf, hw, 1, 2, 2, N, row_groups);
    long long totN2 = (long long)N * 2;
    k_self_init<<<cdiv_u(totN2, B), B, 0, stream>>>(dinv, hw, agg, totN2, 1);
    long long totE1 = (long long)E;
    k_edge_scatter<2><<<cdiv_u(totE1, B), B, 0, stream>>>(src, dst, nrm, hw, agg, totE1, 0, 2);

    // ---- pool (b3 folded post-mean) ----
    k_pool_zero <<<1, 256, 0, stream>>>(pool, cnt);
    k_pool_acc  <<<cdiv_u(N, B), B, 0, stream>>>(agg, batch, pool, cnt, N);
    k_pool_final<<<1, 256, 0, stream>>>(pool, cnt, b3, out);
}
